// GatedAttention_58420145160571
// MI455X (gfx1250) — matrix operand order repaired and checked by reading, not compiled, not run on hardware
//
#include <hip/hip_runtime.h>
#include <math.h>

#define N_NODES  100000
#define N_GRAPHS 64
#define IN_DIM   512
#define EMB      256
#define SPLITS   16

typedef __attribute__((ext_vector_type(16))) __bf16 v16bf;
typedef __attribute__((ext_vector_type(8)))  float  v8f;

__device__ __forceinline__ float sigmoidf_(float v) {
    return 1.0f / (1.0f + __expf(-v));
}

__device__ __forceinline__ void atomicMaxF(float* addr, float val) {
    int* ai = (int*)addr;
    int cur = *ai;
    while (__int_as_float(cur) < val) {
        int prev = atomicCAS(ai, cur, __float_as_int(val));
        if (prev == cur) break;
        cur = prev;
    }
}

// ---------------------------------------------------------------------------
// Init: zero x_graphs output region, init per-segment max/sum accumulators.
// ---------------------------------------------------------------------------
__global__ void kInit(float* __restrict__ xg, float* __restrict__ segmax,
                      float* __restrict__ segsum) {
    int i = blockIdx.x * 256 + threadIdx.x;
    if (i < N_GRAPHS * IN_DIM) xg[i] = 0.0f;
    if (i < N_GRAPHS) { segmax[i] = -INFINITY; segsum[i] = 0.0f; }
}

// ---------------------------------------------------------------------------
// Pack V_w / U_w (fp32, KxN row-major) into bf16 arranged in WMMA B-fragment
// register order: frag[((kt*2+half)*EMB + n)*16 + j] = W[(32kt+16h+j)*EMB + n]
// so each lane reads its whole 16-elem fragment as one 32B contiguous chunk.
// ---------------------------------------------------------------------------
__global__ void kPackW(const float* __restrict__ Vw, const float* __restrict__ Uw,
                       __bf16* __restrict__ pV, __bf16* __restrict__ pU) {
    int idx = blockIdx.x * 256 + threadIdx.x;
    if (idx >= IN_DIM * EMB) return;
    int j  = idx & 15;
    int n  = (idx >> 4) & (EMB - 1);
    int kh = idx >> 12;                          // kt*2 + half, 0..31
    int k  = (kh >> 1) * 32 + (kh & 1) * 16 + j;
    pV[idx] = (__bf16)Vw[k * EMB + n];
    pU[idx] = (__bf16)Uw[k * EMB + n];
}

// ---------------------------------------------------------------------------
// Kernel A: fused dual-GEMM (x@V_w, x@U_w in bf16 WMMA) + gate + dot(w_w)
//           -> per-node score s, per-segment running max.
// One workgroup = 16 node rows. 8 waves, each wave owns 32 emb columns
// for BOTH V and U paths (4 x v8f f32 accumulators).
// ---------------------------------------------------------------------------
__global__ __launch_bounds__(256, 2)
void kA_scores(const float* __restrict__ x, const long long* __restrict__ ptr,
               const __bf16* __restrict__ pV, const __bf16* __restrict__ pU,
               const float* __restrict__ Vb, const float* __restrict__ Ub,
               const float* __restrict__ ww, const float* __restrict__ wb,
               float* __restrict__ s, float* __restrict__ segmax) {
    const int tid  = threadIdx.x;
    const int lane = tid & 31;
    const int wave = tid >> 5;
    const int half = lane >> 4;
    const int l15  = lane & 15;
    const int row0 = blockIdx.x * 16;
    const int arow = row0 + l15;           // A-fragment row for this lane

    __shared__ float sred[16];
    if (tid < 16) sred[tid] = 0.0f;
    __syncthreads();

    const v16bf* pVf = (const v16bf*)pV;
    const v16bf* pUf = (const v16bf*)pU;

    const int nc0 = wave * 32;             // this wave's first output column
    v8f accV0 = {}; v8f accV1 = {};
    v8f accU0 = {}; v8f accU1 = {};

    for (int kt = 0; kt < IN_DIM / 32; ++kt) {
        const int k0 = kt * 32;
        // ---- A fragment: 16x32 bf16. Lane (half,m) holds
        // K = 16*half+[0..16) in element order (matches packed B):
        // one contiguous 64B chunk of the x row -> 4x b128 loads.
        const float* xr = x + (size_t)arow * IN_DIM + k0;
        const float4* c0p = (const float4*)(xr + half * 16);
        const float4* c1p = (const float4*)(xr + half * 16 + 8);
        float4 f0 = c0p[0], f1 = c0p[1];
        float4 f2 = c1p[0], f3 = c1p[1];
        v16bf a;
        a[0]  = (__bf16)f0.x; a[1]  = (__bf16)f0.y;
        a[2]  = (__bf16)f0.z; a[3]  = (__bf16)f0.w;
        a[4]  = (__bf16)f1.x; a[5]  = (__bf16)f1.y;
        a[6]  = (__bf16)f1.z; a[7]  = (__bf16)f1.w;
        a[8]  = (__bf16)f2.x; a[9]  = (__bf16)f2.y;
        a[10] = (__bf16)f2.z; a[11] = (__bf16)f2.w;
        a[12] = (__bf16)f3.x; a[13] = (__bf16)f3.y;
        a[14] = (__bf16)f3.z; a[15] = (__bf16)f3.w;
        if (k0 + 32 < IN_DIM) __builtin_prefetch(xr + 32, 0, 0);

        // ---- B fragments: one 32B contiguous pre-packed chunk per lane.
        const int fbase = (kt * 2 + half) * EMB;
        {
            const int fi = fbase + nc0 + l15;
            v16bf bV = pVf[fi];
            v16bf bU = pUf[fi];
            accV0 = __builtin_amdgcn_wmma_f32_16x16x32_bf16(
                false, a, false, bV, (short)0, accV0, false, false);
            accU0 = __builtin_amdgcn_wmma_f32_16x16x32_bf16(
                false, a, false, bU, (short)0, accU0, false, false);
        }
        {
            const int fi = fbase + nc0 + 16 + l15;
            v16bf bV = pVf[fi];
            v16bf bU = pUf[fi];
            accV1 = __builtin_amdgcn_wmma_f32_16x16x32_bf16(
                false, a, false, bV, (short)0, accV1, false, false);
            accU1 = __builtin_amdgcn_wmma_f32_16x16x32_bf16(
                false, a, false, bU, (short)0, accU1, false, false);
        }
    }

    // ---- Epilogue: gate = tanh(V+Vb)*sigmoid(U+Ub)*w, reduce over columns.
    // C layout: VGPR r -> (M = r + 8*half, N = l15) for column nc0(+16)+l15.
    const int c0 = nc0 + l15, c1 = c0 + 16;
    const float w0  = ww[c0], w1  = ww[c1];
    const float bv0 = Vb[c0], bv1 = Vb[c1];
    const float bu0 = Ub[c0], bu1 = Ub[c1];
#pragma unroll
    for (int r = 0; r < 8; ++r) {
        float g0 = tanhf(accV0[r] + bv0) * sigmoidf_(accU0[r] + bu0) * w0;
        float g1 = tanhf(accV1[r] + bv1) * sigmoidf_(accU1[r] + bu1) * w1;
        float v = g0 + g1;
#pragma unroll
        for (int m = 1; m < 16; m <<= 1) v += __shfl_xor(v, m, 32);
        if (l15 == 0) atomicAdd(&sred[r + 8 * half], v);
    }
    __syncthreads();

    if (tid < 16) {
        const int row = row0 + tid;
        const float sv = sred[tid] + wb[0];
        s[row] = sv;
        // largest g with ptr[g] <= row (may be N_GRAPHS for tail rows)
        int lo = 0, hi = N_GRAPHS;
        while (lo < hi) {
            int mid = (lo + hi + 1) >> 1;
            if ((long long)row >= ptr[mid]) lo = mid; else hi = mid - 1;
        }
        if (lo < N_GRAPHS) atomicMaxF(&segmax[lo], sv);
    }
}

// ---------------------------------------------------------------------------
// Kernel B: per-segment sum of exp(s - max).
// ---------------------------------------------------------------------------
__global__ __launch_bounds__(256)
void kB_segsum(const float* __restrict__ s, const long long* __restrict__ ptr,
               const float* __restrict__ segmax, float* __restrict__ segsum) {
    const int g = blockIdx.x;
    const long long start = ptr[g], end = ptr[g + 1];
    const float mx = segmax[g];
    float loc = 0.0f;
    for (long long i = start + threadIdx.x; i < end; i += 256)
        loc += __expf(s[i] - mx);
    __shared__ float red[256];
    red[threadIdx.x] = loc;
    __syncthreads();
    for (int o = 128; o > 0; o >>= 1) {
        if (threadIdx.x < o) red[threadIdx.x] += red[threadIdx.x + o];
        __syncthreads();
    }
    if (threadIdx.x == 0) segsum[g] = red[0];
}

// ---------------------------------------------------------------------------
// Kernel C: att = softmax, x_graphs = segment_sum(att * x).
// grid = 64 segments x SPLITS row-chunks; thread owns columns c and c+256.
// ---------------------------------------------------------------------------
__global__ __launch_bounds__(256)
void kC_pool(const float* __restrict__ x, const long long* __restrict__ ptr,
             const float* __restrict__ s, const float* __restrict__ segmax,
             const float* __restrict__ segsum,
             float* __restrict__ att, float* __restrict__ xg) {
    const int g  = blockIdx.x / SPLITS;
    const int sp = blockIdx.x % SPLITS;
    const long long start = ptr[g], end = ptr[g + 1];
    const long long total = end - start;
    const long long chunk = (total + SPLITS - 1) / SPLITS;
    long long r0 = start + (long long)sp * chunk;
    long long r1 = r0 + chunk; if (r1 > end) r1 = end;

    const float mx  = segmax[g];
    const float inv = 1.0f / segsum[g];
    const int c = threadIdx.x;
    float acc0 = 0.0f, acc1 = 0.0f;
    for (long long i = r0; i < r1; ++i) {
        const float a = __expf(s[i] - mx) * inv;
        if (c == 0) att[i] = a;
        const float* xr = x + (size_t)i * IN_DIM;
        acc0 += a * xr[c];
        acc1 += a * xr[c + 256];
    }
    if (r0 < r1) {
        atomicAdd(&xg[g * IN_DIM + c], acc0);
        atomicAdd(&xg[g * IN_DIM + c + 256], acc1);
    }
}

// ---------------------------------------------------------------------------
// Kernel D: tail nodes past ptr[64] (reference's out-of-range segment -> JAX
// clamps gathers to segment 63; they contribute to no pooled graph).
// ---------------------------------------------------------------------------
__global__ void kD_tail(const float* __restrict__ s,
                        const long long* __restrict__ ptr,
                        const float* __restrict__ segmax,
                        const float* __restrict__ segsum,
                        float* __restrict__ att) {
    const long long start = ptr[N_GRAPHS];
    const long long i = start + threadIdx.x + (long long)blockIdx.x * blockDim.x;
    if (i < N_NODES)
        att[i] = __expf(s[i] - segmax[N_GRAPHS - 1]) / segsum[N_GRAPHS - 1];
}

extern "C" void kernel_launch(void* const* d_in, const int* in_sizes, int n_in,
                              void* d_out, int out_size, void* d_ws, size_t ws_size,
                              hipStream_t stream) {
    const float*     x   = (const float*)d_in[0];
    const long long* ptr = (const long long*)d_in[1];
    const float*     Vw  = (const float*)d_in[2];
    const float*     Vb  = (const float*)d_in[3];
    const float*     Uw  = (const float*)d_in[4];
    const float*     Ub  = (const float*)d_in[5];
    const float*     ww  = (const float*)d_in[6];
    const float*     wb  = (const float*)d_in[7];

    float* att = (float*)d_out;                 // [N_NODES]
    float* xg  = (float*)d_out + N_NODES;       // [N_GRAPHS * IN_DIM]

    float* s      = (float*)d_ws;               // [N_NODES]
    float* segmax = s + N_NODES;                // [N_GRAPHS]
    float* segsum = segmax + N_GRAPHS;          // [N_GRAPHS]
    // packed bf16 weights, 32B-aligned region after the fp32 scratch
    __bf16* pV = (__bf16*)((char*)d_ws + 401408);            // 256 KB
    __bf16* pU = (__bf16*)((char*)d_ws + 401408 + 262144);   // 256 KB

    kInit<<<(N_GRAPHS * IN_DIM + 255) / 256, 256, 0, stream>>>(xg, segmax, segsum);
    kPackW<<<(IN_DIM * EMB + 255) / 256, 256, 0, stream>>>(Vw, Uw, pV, pU);
    kA_scores<<<N_NODES / 16, 256, 0, stream>>>(x, ptr, pV, pU, Vb, Ub, ww, wb,
                                                s, segmax);
    kB_segsum<<<N_GRAPHS, 256, 0, stream>>>(s, ptr, segmax, segsum);
    kC_pool<<<N_GRAPHS * SPLITS, 256, 0, stream>>>(x, ptr, s, segmax, segsum,
                                                   att, xg);
    kD_tail<<<1, 64, 0, stream>>>(s, ptr, segmax, segsum, att);
}
